// CRF_72954314490472
// MI455X (gfx1250) — matrix operand order repaired and checked by reading, not compiled, not run on hardware
//
#include <hip/hip_runtime.h>
#include <hip/hip_bf16.h>
#include <stdint.h>

#define SEQ_LEN  1024
#define BATCH    512
#define NUM_TAGS 64

#define INV_LN2 1.4426950408889634f
#define LN2     0.6931471805599453f

typedef __attribute__((ext_vector_type(16))) _Float16 v16h;
typedef __attribute__((ext_vector_type(8)))  _Float16 v8h;
typedef __attribute__((ext_vector_type(8)))  float    v8f;
typedef __attribute__((ext_vector_type(4)))  float    v4f;

// ---------------------------------------------------------------------------
// K1: E[i][j] = exp(transitions[i][j]) in f16 (feeds WMMA B-frags)
// ---------------------------------------------------------------------------
__global__ __launch_bounds__(256)
void crf_exp_trans_kernel(const float* __restrict__ trans, _Float16* __restrict__ E) {
    int i = blockIdx.x * blockDim.x + threadIdx.x;
    if (i < NUM_TAGS * NUM_TAGS) E[i] = (_Float16)__expf(trans[i]);
}

// ---------------------------------------------------------------------------
// K2: forward log-partition scan in log2 domain, ONE barrier per step.
//     Workgroup = 4 waves = one 16-batch-row group; wave w owns tag tile
//     [16w, 16w+16). Each wave centers its p-slice by its OWN partial max
//     (exp runs pre-barrier); readers rescale A-fragment K-groups by
//     2^(M_w' - M) after the single barrier. Double-buffered LDS.
// ---------------------------------------------------------------------------
__global__ __launch_bounds__(128)
void crf_forward_kernel(const float* __restrict__ em,
                        const uint8_t* __restrict__ mask,
                        const _Float16* __restrict__ E,
                        const float* __restrict__ startT,
                        const float* __restrict__ endT,
                        float* __restrict__ logZ) {
    __shared__ _Float16 sA[2][16 * NUM_TAGS]; // 2x2 KB slice-centered p-tile
    __shared__ float    smax[2][16 * 4];      // 2x per-row partial maxes [m][w]
    __shared__ float    ssum[16 * 4];         // per-row partial sums [m][w]

    const int tid  = threadIdx.x;
    const int w    = tid >> 5;         // wave id 0..3  == N-tile index
    const int lane = tid & 31;
    const int half = lane >> 4;        // 0 or 1
    const int lm   = lane & 15;
    const int bb   = blockIdx.x * 16;  // first batch row of this workgroup
    const int tag  = 16 * w + lm;      // this lane's tag column

    // --- B fragments for this wave's N-tile: 2 K-chunks of 32x16.
    // CDNA5 f16 B layout (32x16): lane -> column N=lm, element e -> K = 8*half + (e&7) + 16*(e>>3).
    v16h bfrag[2];
    for (int k2 = 0; k2 < 2; ++k2) {
        v16h f;
        #pragma unroll
        for (int e = 0; e < 16; ++e) {
            int K = 32 * k2 + 8 * half + (e & 7) + 16 * (e >> 3);
            f[e] = E[K * NUM_TAGS + tag];
        }
        bfrag[k2] = f;
    }

    // --- L (= score / ln2), C/D layout: sc[r] <-> (batch m = r + 8*half, tag)
    float sc[8];
    {
        float st = startT[tag];
        #pragma unroll
        for (int r = 0; r < 8; ++r) {
            int b = bb + r + 8 * half;
            sc[r] = (st + em[(size_t)b * NUM_TAGS + tag]) * INV_LN2;
        }
    }

    // loop-carried per-lane pointers; loads use constant instruction offsets.
    const float* emp = em + (size_t)BATCH * NUM_TAGS
                          + (size_t)(bb + 8 * half) * NUM_TAGS + tag;
    const uint8_t* mkp = mask + BATCH + bb + 8 * half;

    for (int t = 1; t < SEQ_LEN; ++t) {
        const int pb = t & 1;

        // ---- issue this step's global loads first (latency hidden below)
        float emt[8];
        float mf[8];
        #pragma unroll
        for (int r = 0; r < 8; ++r) emt[r] = emp[r * NUM_TAGS];
        #pragma unroll
        for (int r = 0; r < 8; ++r) mf[r] = (float)mkp[r];
        emp += (size_t)BATCH * NUM_TAGS;
        mkp += BATCH;

        // ---- partial row max over this wave's 16 tags, then slice-centered
        //      p = 2^(L - M_w) straight into the shared p-tile (pre-barrier!)
        #pragma unroll
        for (int r = 0; r < 8; ++r) {
            float v = sc[r];
            v = fmaxf(v, __shfl_xor(v, 1, 32));
            v = fmaxf(v, __shfl_xor(v, 2, 32));
            v = fmaxf(v, __shfl_xor(v, 4, 32));
            v = fmaxf(v, __shfl_xor(v, 8, 32));
            if (lm == 0) smax[pb][(r + 8 * half) * 4 + w] = v;
            sA[pb][(r + 8 * half) * NUM_TAGS + tag] =
                (_Float16)__builtin_amdgcn_exp2f(sc[r] - v);
        }
        __syncthreads();   // the ONLY barrier in the steady-state loop

        // ---- A-row rescale factors: row of A = lane lm; fac[w'] = 2^(M_w' - M)
        v4f pmL = *(const v4f*)&smax[pb][lm * 4];
        float ML = fmaxf(fmaxf(pmL.x, pmL.y), fmaxf(pmL.z, pmL.w));
        _Float16 fac[4];
        fac[0] = (_Float16)__builtin_amdgcn_exp2f(pmL.x - ML);
        fac[1] = (_Float16)__builtin_amdgcn_exp2f(pmL.y - ML);
        fac[2] = (_Float16)__builtin_amdgcn_exp2f(pmL.z - ML);
        fac[3] = (_Float16)__builtin_amdgcn_exp2f(pmL.w - ML);

        // ---- A fragments (16x32 f16), then per-K-group rescale.
        //      chunk c, elems 0..7 -> K-tile 2c ; elems 8..15 -> K-tile 2c+1.
        v16h afrag[2];
        #pragma unroll
        for (int c = 0; c < 2; ++c) {
            int baseLo = 32 * c + 8 * half;
            v8h lo = *(const v8h*)&sA[pb][lm * NUM_TAGS + baseLo];
            v8h hi = *(const v8h*)&sA[pb][lm * NUM_TAGS + baseLo + 16];
            v16h f = __builtin_shufflevector(lo, hi,
                        0,1,2,3,4,5,6,7,8,9,10,11,12,13,14,15);
            #pragma unroll
            for (int e = 0; e < 8; ++e)  f[e] = f[e] * fac[2 * c];
            #pragma unroll
            for (int e = 8; e < 16; ++e) f[e] = f[e] * fac[2 * c + 1];
            afrag[c] = f;
        }

        // ---- acc[m][j] = sum_i 2^(L_i[m]-M[m]) * E[i][j] for this wave's tags
        v8f z = {};
        v8f acc = __builtin_amdgcn_wmma_f32_16x16x32_f16(
                      false, afrag[0], false, bfrag[0], (short)0, z, false, false);
        acc = __builtin_amdgcn_wmma_f32_16x16x32_f16(
                      false, afrag[1], false, bfrag[1], (short)0, acc, false, false);

        // ---- global row max for the D-rows, branchless masked update
        #pragma unroll
        for (int r = 0; r < 8; ++r) {
            int m = r + 8 * half;
            v4f pm = *(const v4f*)&smax[pb][m * 4];
            float mrow = fmaxf(fmaxf(pm.x, pm.y), fmaxf(pm.z, pm.w));
            float nxt = fmaf(emt[r], INV_LN2,
                             mrow + __builtin_amdgcn_logf(acc[r]));
            sc[r] = fmaf(mf[r], nxt - sc[r], sc[r]);
        }
        // no trailing barrier: next step writes the other LDS buffer; reuse of
        // this buffer at t+2 is fenced by the step-(t+1) barrier.
    }

    // ---- finalize: logZ[b] = ln2 * log2sumexp2_j(L[b][j] + endT[j]/ln2)
    {
        float e = endT[tag] * INV_LN2;
        #pragma unroll
        for (int r = 0; r < 8; ++r) sc[r] += e;
    }
    __syncthreads();
    #pragma unroll
    for (int r = 0; r < 8; ++r) {
        float v = sc[r];
        v = fmaxf(v, __shfl_xor(v, 1, 32));
        v = fmaxf(v, __shfl_xor(v, 2, 32));
        v = fmaxf(v, __shfl_xor(v, 4, 32));
        v = fmaxf(v, __shfl_xor(v, 8, 32));
        if (lm == 0) smax[0][(r + 8 * half) * 4 + w] = v;
    }
    __syncthreads();
    float mrow[8];
    #pragma unroll
    for (int r = 0; r < 8; ++r) {
        int m = r + 8 * half;
        v4f pm = *(const v4f*)&smax[0][m * 4];
        mrow[r] = fmaxf(fmaxf(pm.x, pm.y), fmaxf(pm.z, pm.w));
        float s = __builtin_amdgcn_exp2f(sc[r] - mrow[r]);
        s += __shfl_xor(s, 1, 32);
        s += __shfl_xor(s, 2, 32);
        s += __shfl_xor(s, 4, 32);
        s += __shfl_xor(s, 8, 32);
        if (lm == 0) ssum[m * 4 + w] = s;
    }
    __syncthreads();
    if (w == 0 && lm == 0) {
        #pragma unroll
        for (int r = 0; r < 8; ++r) {
            int m = r + 8 * half;
            v4f ps = *(const v4f*)&ssum[m * 4];
            float s = (ps.x + ps.y) + (ps.z + ps.w);
            logZ[bb + m] = LN2 * (mrow[r] + __builtin_amdgcn_logf(s));
        }
    }
}

// ---------------------------------------------------------------------------
// K3: gold-path score. Thread-per-batch: tags/mask reads fully coalesced.
// ---------------------------------------------------------------------------
__global__ __launch_bounds__(256)
void crf_score_kernel(const float* __restrict__ em, const int* __restrict__ tags,
                      const uint8_t* __restrict__ mask, const float* __restrict__ trans,
                      const float* __restrict__ startT, const float* __restrict__ endT,
                      float* __restrict__ scoreS) {
    const int b = blockIdx.x * 256 + threadIdx.x;

    int   prev = tags[b];
    float sc   = startT[prev] + em[(size_t)b * NUM_TAGS + prev];
    int   len  = mask[b] ? 1 : 0;

    #pragma unroll 4
    for (int t = 1; t < SEQ_LEN; ++t) {
        int   tc = tags[t * BATCH + b];
        uint8_t mk = mask[t * BATCH + b];
        float mm = (float)mk;
        len += mk ? 1 : 0;
        sc += mm * (trans[prev * NUM_TAGS + tc]
                    + em[(size_t)t * BATCH * NUM_TAGS + (size_t)b * NUM_TAGS + tc]);
        prev = tc;
    }
    int lastT = tags[(size_t)(len - 1) * BATCH + b];
    scoreS[b] = sc + endT[lastT];
}

// ---------------------------------------------------------------------------
// K4: out = mean_b(logZ[b] - scoreS[b])
// ---------------------------------------------------------------------------
__global__ __launch_bounds__(256)
void crf_reduce_kernel(const float* __restrict__ logZ, const float* __restrict__ scoreS,
                       float* __restrict__ out) {
    __shared__ float red[256];
    int i = threadIdx.x;
    red[i] = (logZ[i] - scoreS[i]) + (logZ[i + 256] - scoreS[i + 256]);
    __syncthreads();
    for (int s = 128; s > 0; s >>= 1) {
        if (i < s) red[i] += red[i + s];
        __syncthreads();
    }
    if (i == 0) out[0] = red[0] / (float)BATCH;
}

// ---------------------------------------------------------------------------
extern "C" void kernel_launch(void* const* d_in, const int* in_sizes, int n_in,
                              void* d_out, int out_size, void* d_ws, size_t ws_size,
                              hipStream_t stream) {
    const float*   em     = (const float*)d_in[0];
    const int*     tags   = (const int*)d_in[1];
    const uint8_t* mask   = (const uint8_t*)d_in[2];
    const float*   trans  = (const float*)d_in[3];
    const float*   startT = (const float*)d_in[4];
    const float*   endT   = (const float*)d_in[5];

    char*     ws     = (char*)d_ws;
    _Float16* E      = (_Float16*)ws;                    // 64*64*2 = 8 KB
    float*    logZ   = (float*)(ws + 8192);              // 512*4   = 2 KB
    float*    scoreS = (float*)(ws + 8192 + 2048);       // 512*4   = 2 KB

    crf_exp_trans_kernel<<<16, 256, 0, stream>>>(trans, E);
    crf_forward_kernel<<<BATCH / 16, 128, 0, stream>>>(em, mask, E, startT, endT, logZ);
    crf_score_kernel<<<BATCH / 256, 256, 0, stream>>>(em, tags, mask, trans, startT, endT, scoreS);
    crf_reduce_kernel<<<1, 256, 0, stream>>>(logZ, scoreS, (float*)d_out);
}
